// Unfold3D_10445360464000
// MI455X (gfx1250) — matrix operand order repaired and checked by reading, not compiled, not run on hardware
//
#include <hip/hip_runtime.h>
#include <math.h>

// ---------------------------------------------------------------------------
// Performer block on 3D-unfolded patches, CDNA5 (gfx1250, wave32, WMMA).
// B=4, L=48^3=110592 tokens, DIM=EMB=64, M=32 random features.
// All GEMMs run on v_wmma_f32_16x16x32_f16 (f16 operands, f32 accumulate).
// kp/qp get a common x4 scale (cancels exactly in the attention ratio) so the
// exp() features sit in f16 normal range.
// ---------------------------------------------------------------------------

typedef __attribute__((ext_vector_type(16))) _Float16 v16h;
typedef __attribute__((ext_vector_type(8)))  float    v8f;

#define BATCH      4
#define LTOK       110592
#define TILES_PB   6912            // LTOK / 16
#define SCALE_P    4.0f            // common scale on kp & qp (cancels in ratio)

// ws layout (floats)
#define KPTV_OFF   0               // 4 * 2048  (kptv[b][e][m], e-major)
#define KSUM_OFF   8192            // 4 * 32
#define PART_OFF   8320            // partials: 4 * 384 * 2080
#define K1_BLK_PB  96
#define K1_WPB     (K1_BLK_PB * 4) // waves per batch in k1 = 384
#define PART_STR   2080
#define K2_BLK_PB  432
#define K2_WPB     (K2_BLK_PB * 4) // 1728

#define LDSF() asm volatile("s_wait_dscnt 0" ::: "memory")

__device__ __forceinline__ v8f wmma32(v16h a, v16h b, v8f c) {
  return __builtin_amdgcn_wmma_f32_16x16x32_f16(
      /*neg_a=*/false, a, /*neg_b=*/false, b,
      /*c_mod=*/(short)0, c, /*reuse_a=*/false, /*reuse_b=*/false);
}

// Stage a B fragment (32x16, K x N) into LDS, one v16h per lane.
// Element (k, n) = base[n * sn + k * sk]; lane = n + 16*((k>>3)&1), half = (k&7) + 8*(k>=16).
__device__ __forceinline__ void stageB(_Float16* frag, const float* base,
                                       int sn, int sk, int tid, int nthr) {
  for (int idx = tid; idx < 512; idx += nthr) {
    const int lane = idx >> 4, e = idx & 15;
    const int n = lane & 15;
    const int k = ((lane >> 4) << 3) + (e & 7) + ((e >> 3) << 4);
    frag[(lane << 4) + e] = (_Float16)base[n * sn + k * sk];
  }
}

__device__ __forceinline__ v16h loadB(const _Float16* frag, int lane) {
  union { v16h v; uint4 u[2]; } r;
  const uint4* p = (const uint4*)(frag + (lane << 4));
  r.u[0] = p[0]; r.u[1] = p[1];
  return r.v;
}

// A fragment (16x32 f16) from a row-major 16x64 f16 LDS buffer.
// lane: row m = lane&15;  half e -> k = kc*32 + (e>>3)*16 + (lane>>4)*8 + (e&7)
// => two contiguous 16-byte LDS reads per lane.
__device__ __forceinline__ v16h loadA(const _Float16* buf, int lane, int kc) {
  const int m  = lane & 15;
  const int kb = kc * 32 + ((lane >> 4) << 3);
  union { v16h v; uint4 u[2]; } r;
  r.u[0] = *(const uint4*)(buf + m * 64 + kb);
  r.u[1] = *(const uint4*)(buf + m * 64 + kb + 16);
  return r.v;
}

// Unfold (4,1,96^3) -> 16-token tile (16x64), then LayerNorm -> f16 A source.
// Lane pair per token: lane handles features (lane&1)*32 .. +31.
__device__ __forceinline__ void unfold_ln(
    const float* __restrict__ x, const float* __restrict__ g1,
    const float* __restrict__ b1, int b, int l0, int lane,
    float* Sw, _Float16* Hw) {
  const int t  = lane >> 1;
  const int fb = (lane & 1) * 32;
  const int ltk = l0 + t;
  const int dz = ltk / 2304;
  const int hy = (ltk / 48) % 48;
  const int wx = ltk % 48;
  const size_t xbase = (size_t)b * 884736;
  float sm = 0.f, sq = 0.f;
  for (int j = 0; j < 32; ++j) {
    const int f  = fb + j;
    const int id = 2 * dz - 1 + (f >> 4);
    const int ih = 2 * hy - 1 + ((f >> 2) & 3);
    const int iw = 2 * wx - 1 + (f & 3);
    float v = 0.f;
    if ((unsigned)id < 96u && (unsigned)ih < 96u && (unsigned)iw < 96u)
      v = x[xbase + ((size_t)id * 96 + ih) * 96 + iw];
    Sw[t * 65 + f] = v; sm += v; sq += v * v;
  }
  sm += __shfl_xor(sm, 1);
  sq += __shfl_xor(sq, 1);
  const float mu = sm * (1.f / 64.f);
  const float rs = rsqrtf(sq * (1.f / 64.f) - mu * mu + 1e-5f);
  for (int j = 0; j < 32; ++j) {
    const int f = fb + j;
    Hw[t * 64 + f] = (_Float16)((Sw[t * 65 + f] - mu) * rs * g1[f] + b1[f]);
  }
}

// ---------------------------------------------------------------------------
// Kernel 1: per-batch reductions ksum (32) and kptv (64x32) -> per-wave
// partials in ws (deterministic, no atomics).
// ---------------------------------------------------------------------------
__global__ __launch_bounds__(128) void k1_reduce(
    const float* __restrict__ x, const float* __restrict__ g1,
    const float* __restrict__ b1, const float* __restrict__ wkqv,
    const float* __restrict__ bkqv, const float* __restrict__ wrf,
    float* __restrict__ ws) {
  // B fragments: 0..7 Wk (nt*2+kc), 8..15 Wv, 16..19 w_rf
  __shared__ __align__(16) _Float16 WF[20][512];
  __shared__ float S [4][16][65];
  __shared__ float VB[4][16][65];
  __shared__ __align__(16) _Float16 H16[4][16 * 64];
  __shared__ float XD[4][16];

  const int tid = threadIdx.x;
  for (int nt = 0; nt < 4; ++nt)
    for (int kc = 0; kc < 2; ++kc) {
      stageB(WF[nt * 2 + kc],     wkqv + (nt * 16) * 64 + kc * 32,       64, 1, tid, 128); // k
      stageB(WF[8 + nt * 2 + kc], wkqv + (128 + nt * 16) * 64 + kc * 32, 64, 1, tid, 128); // v
    }
  for (int nt = 0; nt < 2; ++nt)
    for (int kc = 0; kc < 2; ++kc)
      stageB(WF[16 + nt * 2 + kc], wrf + (nt * 16) * 64 + kc * 32, 64, 1, tid, 128);
  __syncthreads();

  const int wv = tid >> 5, lane = tid & 31;
  const int b  = blockIdx.x / K1_BLK_PB;
  const int wg = (blockIdx.x % K1_BLK_PB) * 4 + wv;
  float* Sw = &S [wv][0][0];
  float* Vw = &VB[wv][0][0];
  _Float16* Hw = H16[wv];
  const int t = lane >> 1, fb = (lane & 1) * 32;

  v8f acc[8];                       // kptv accumulators: [etile*2 + mtile]
  for (int i = 0; i < 8; ++i)
    for (int r = 0; r < 8; ++r) acc[i][r] = 0.f;
  float ksum_acc = 0.f;             // m = lane

  for (int tile = wg; tile < TILES_PB; tile += K1_WPB) {
    const int l0 = tile * 16;
    unfold_ln(x, g1, b1, b, l0, lane, Sw, Hw);
    LDSF();
    {
      const v16h a0 = loadA(Hw, lane, 0), a1 = loadA(Hw, lane, 1);
      const int n = lane & 15, mb = (lane >> 4) * 8;
      for (int nt = 0; nt < 4; ++nt) {               // v = h @ Wv^T
        v8f c; for (int r = 0; r < 8; ++r) c[r] = 0.f;
        c = wmma32(a0, loadB(WF[8 + nt * 2 + 0], lane), c);
        c = wmma32(a1, loadB(WF[8 + nt * 2 + 1], lane), c);
        const int nG = nt * 16 + n; const float bs = bkqv[128 + nG];
        for (int r = 0; r < 8; ++r) Vw[(mb + r) * 65 + nG] = c[r] + bs;
      }
      for (int nt = 0; nt < 4; ++nt) {               // k = h @ Wk^T
        v8f c; for (int r = 0; r < 8; ++r) c[r] = 0.f;
        c = wmma32(a0, loadB(WF[nt * 2 + 0], lane), c);
        c = wmma32(a1, loadB(WF[nt * 2 + 1], lane), c);
        const int nG = nt * 16 + n; const float bs = bkqv[nG];
        for (int r = 0; r < 8; ++r) Sw[(mb + r) * 65 + nG] = c[r] + bs;
      }
    }
    LDSF();
    {                                                // xd = 0.5*||k||^2 ; k -> f16
      float s2 = 0.f;
      for (int j = 0; j < 32; ++j) { const float kv = Sw[t * 65 + fb + j]; s2 += kv * kv; }
      s2 += __shfl_xor(s2, 1);
      XD[wv][t] = 0.5f * s2;
      for (int j = 0; j < 32; ++j) Hw[t * 64 + fb + j] = (_Float16)Sw[t * 65 + fb + j];
    }
    LDSF();
    {                                                // wtx = k @ w_rf^T  (16x32)
      const v16h a0 = loadA(Hw, lane, 0), a1 = loadA(Hw, lane, 1);
      const int n = lane & 15, mb = (lane >> 4) * 8;
      for (int nt = 0; nt < 2; ++nt) {
        v8f c; for (int r = 0; r < 8; ++r) c[r] = 0.f;
        c = wmma32(a0, loadB(WF[16 + nt * 2 + 0], lane), c);
        c = wmma32(a1, loadB(WF[16 + nt * 2 + 1], lane), c);
        const int nG = nt * 16 + n;
        for (int r = 0; r < 8; ++r) Sw[(mb + r) * 65 + nG] = c[r];
      }
    }
    LDSF();
    {                                                // kp = exp(wtx - xd)*SCALE
      const int m = lane;                            // lane owns feature column m
      for (int tt = 0; tt < 16; ++tt) {
        const float p = expf(Sw[tt * 65 + m] - XD[wv][tt]) * SCALE_P;
        Sw[tt * 65 + m] = p;
        ksum_acc += p;
      }
    }
    LDSF();
    {                                                // kptv += v^T(16e x 16t) @ kp(16t x 16m)
      v16h bk[2];
      for (int mt = 0; mt < 2; ++mt) {               // B: k=t (pad 32), n=m
        union { v16h v; _Float16 h[16]; } r;
        const int m = mt * 16 + (lane & 15);
        for (int e = 0; e < 8; ++e)  r.h[e] = (_Float16)Sw[(((lane >> 4) << 3) + e) * 65 + m];
        for (int e = 8; e < 16; ++e) r.h[e] = (_Float16)0.f;
        bk[mt] = r.v;
      }
      for (int et = 0; et < 4; ++et) {               // A: row=e, k=t (from v, transposed)
        union { v16h v; _Float16 h[16]; } a;
        const int eg = et * 16 + (lane & 15);
        const int kb = (lane >> 4) * 8;
        for (int h2 = 0; h2 < 8;  ++h2) a.h[h2] = (_Float16)Vw[(kb + h2) * 65 + eg];
        for (int h2 = 8; h2 < 16; ++h2) a.h[h2] = (_Float16)0.f;
        acc[et * 2 + 0] = wmma32(a.v, bk[0], acc[et * 2 + 0]);
        acc[et * 2 + 1] = wmma32(a.v, bk[1], acc[et * 2 + 1]);
      }
    }
    LDSF();
  }
  // write this wave's partial (2048 kptv + 32 ksum floats)
  float* pb = ws + PART_OFF + ((size_t)(b * K1_WPB + wg)) * PART_STR;
  const int n = lane & 15, mb = (lane >> 4) * 8;
  for (int et = 0; et < 4; ++et)
    for (int mt = 0; mt < 2; ++mt)
      for (int r = 0; r < 8; ++r)
        pb[(et * 16 + mb + r) * 32 + mt * 16 + n] = acc[et * 2 + mt][r];
  pb[2048 + lane] = ksum_acc;
}

// ---------------------------------------------------------------------------
// Kernel 1b: deterministic serial reduction of the per-wave partials.
// ---------------------------------------------------------------------------
__global__ void k_red(float* __restrict__ ws) {
  const int idx = blockIdx.x * blockDim.x + threadIdx.x;
  if (idx >= BATCH * PART_STR) return;
  const int b = idx / PART_STR, j = idx % PART_STR;
  const float* pb = ws + PART_OFF + (size_t)b * K1_WPB * PART_STR;
  float s = 0.f;
  for (int w = 0; w < K1_WPB; ++w) s += pb[(size_t)w * PART_STR + j];
  if (j < 2048) ws[KPTV_OFF + b * 2048 + j] = s;
  else          ws[KSUM_OFF + b * 32 + (j - 2048)] = s;
}

// ---------------------------------------------------------------------------
// Kernel 2: recompute per-tile q,v; qp; D; attention; proj(+v); LN2; MLP; out.
// ---------------------------------------------------------------------------
__global__ __launch_bounds__(128) void k2_finalize(
    const float* __restrict__ x, const float* __restrict__ g1,
    const float* __restrict__ b1, const float* __restrict__ wkqv,
    const float* __restrict__ bkqv, const float* __restrict__ wproj,
    const float* __restrict__ bproj, const float* __restrict__ g2,
    const float* __restrict__ b2, const float* __restrict__ wfc1,
    const float* __restrict__ bfc1, const float* __restrict__ wfc2,
    const float* __restrict__ bfc2, const float* __restrict__ wrf,
    const float* __restrict__ ws, float* __restrict__ out) {
  // 0..7 Wq, 8..15 Wv, 16..19 w_rf, 20..27 proj, 28..35 fc1, 36..43 fc2, 44..47 kptvB
  __shared__ __align__(16) _Float16 WF[48][512];
  __shared__ float S [4][16][65];
  __shared__ float VB[4][16][65];
  __shared__ float YB[4][16][65];
  __shared__ __align__(16) _Float16 H16[4][16 * 64];
  __shared__ float XD[4][16];
  __shared__ float DV[4][16];
  __shared__ float KS[32];

  const int tid = threadIdx.x;
  const int b   = blockIdx.x / K2_BLK_PB;
  for (int nt = 0; nt < 4; ++nt)
    for (int kc = 0; kc < 2; ++kc) {
      stageB(WF[nt * 2 + kc],      wkqv + (64 + nt * 16) * 64 + kc * 32,  64, 1, tid, 128); // q
      stageB(WF[8 + nt * 2 + kc],  wkqv + (128 + nt * 16) * 64 + kc * 32, 64, 1, tid, 128); // v
      stageB(WF[20 + nt * 2 + kc], wproj + (nt * 16) * 64 + kc * 32,      64, 1, tid, 128);
      stageB(WF[28 + nt * 2 + kc], wfc1 + (nt * 16) * 64 + kc * 32,       64, 1, tid, 128);
      stageB(WF[36 + nt * 2 + kc], wfc2 + (nt * 16) * 64 + kc * 32,       64, 1, tid, 128);
    }
  for (int nt = 0; nt < 2; ++nt)
    for (int kc = 0; kc < 2; ++kc)
      stageB(WF[16 + nt * 2 + kc], wrf + (nt * 16) * 64 + kc * 32, 64, 1, tid, 128);
  for (int nt = 0; nt < 4; ++nt)   // B[k=m][n=e] = kptv[b][e][m]
    stageB(WF[44 + nt], ws + KPTV_OFF + b * 2048 + nt * 16 * 32, 32, 1, tid, 128);
  if (tid < 32) KS[tid] = ws[KSUM_OFF + b * 32 + tid];
  __syncthreads();

  const int wv = tid >> 5, lane = tid & 31;
  const int wg = (blockIdx.x % K2_BLK_PB) * 4 + wv;
  float* Sw = &S [wv][0][0];
  float* Vw = &VB[wv][0][0];
  float* Yw = &YB[wv][0][0];
  _Float16* Hw = H16[wv];
  const int t = lane >> 1, fb = (lane & 1) * 32;
  const int n = lane & 15, mb = (lane >> 4) * 8;

  for (int tile = wg; tile < TILES_PB; tile += K2_WPB) {
    const int l0 = tile * 16;
    unfold_ln(x, g1, b1, b, l0, lane, Sw, Hw);
    LDSF();
    {
      const v16h a0 = loadA(Hw, lane, 0), a1 = loadA(Hw, lane, 1);
      for (int nt = 0; nt < 4; ++nt) {               // v -> VB
        v8f c; for (int r = 0; r < 8; ++r) c[r] = 0.f;
        c = wmma32(a0, loadB(WF[8 + nt * 2 + 0], lane), c);
        c = wmma32(a1, loadB(WF[8 + nt * 2 + 1], lane), c);
        const int nG = nt * 16 + n; const float bs = bkqv[128 + nG];
        for (int r = 0; r < 8; ++r) Vw[(mb + r) * 65 + nG] = c[r] + bs;
      }
      for (int nt = 0; nt < 4; ++nt) {               // q -> S
        v8f c; for (int r = 0; r < 8; ++r) c[r] = 0.f;
        c = wmma32(a0, loadB(WF[nt * 2 + 0], lane), c);
        c = wmma32(a1, loadB(WF[nt * 2 + 1], lane), c);
        const int nG = nt * 16 + n; const float bs = bkqv[64 + nG];
        for (int r = 0; r < 8; ++r) Sw[(mb + r) * 65 + nG] = c[r] + bs;
      }
    }
    LDSF();
    {                                                // xd ; q -> f16
      float s2 = 0.f;
      for (int j = 0; j < 32; ++j) { const float qv = Sw[t * 65 + fb + j]; s2 += qv * qv; }
      s2 += __shfl_xor(s2, 1);
      XD[wv][t] = 0.5f * s2;
      for (int j = 0; j < 32; ++j) Hw[t * 64 + fb + j] = (_Float16)Sw[t * 65 + fb + j];
    }
    LDSF();
    {                                                // wtx = q @ w_rf^T
      const v16h a0 = loadA(Hw, lane, 0), a1 = loadA(Hw, lane, 1);
      for (int nt = 0; nt < 2; ++nt) {
        v8f c; for (int r = 0; r < 8; ++r) c[r] = 0.f;
        c = wmma32(a0, loadB(WF[16 + nt * 2 + 0], lane), c);
        c = wmma32(a1, loadB(WF[16 + nt * 2 + 1], lane), c);
        const int nG = nt * 16 + n;
        for (int r = 0; r < 8; ++r) Sw[(mb + r) * 65 + nG] = c[r];
      }
    }
    LDSF();
    {                                                // qp = exp(wtx - xd)*SCALE
      const int m = lane;
      for (int tt = 0; tt < 16; ++tt) {
        const float p = expf(Sw[tt * 65 + m] - XD[wv][tt]) * SCALE_P;
        Sw[tt * 65 + m] = p;
        Hw[tt * 64 + m] = (_Float16)p;               // qp f16 in cols 0..31
      }
    }
    LDSF();
    {                                                // D[t] = sum_m qp*ksum
      const int ms = (lane & 1) * 16;
      float d = 0.f;
      for (int j = 0; j < 16; ++j) d += Sw[t * 65 + ms + j] * KS[ms + j];
      d += __shfl_xor(d, 1);
      DV[wv][t] = d;
    }
    LDSF();
    {                                                // y_att = qp @ kptv^T, /D -> Hw f16
      const v16h aq = loadA(Hw, lane, 0);            // qp is 16x32 (kc=0 only)
      for (int nt = 0; nt < 4; ++nt) {
        v8f c; for (int r = 0; r < 8; ++r) c[r] = 0.f;
        c = wmma32(aq, loadB(WF[44 + nt], lane), c);
        const int nG = nt * 16 + n;
        for (int r = 0; r < 8; ++r)
          Hw[(mb + r) * 64 + nG] = (_Float16)(c[r] / (DV[wv][mb + r] + 1e-8f));
      }
    }
    LDSF();
    {                                                // y = v + y_att @ Wp^T + bp -> YB
      const v16h a0 = loadA(Hw, lane, 0), a1 = loadA(Hw, lane, 1);
      for (int nt = 0; nt < 4; ++nt) {
        v8f c; for (int r = 0; r < 8; ++r) c[r] = 0.f;
        c = wmma32(a0, loadB(WF[20 + nt * 2 + 0], lane), c);
        c = wmma32(a1, loadB(WF[20 + nt * 2 + 1], lane), c);
        const int nG = nt * 16 + n; const float bs = bproj[nG];
        for (int r = 0; r < 8; ++r)
          Yw[(mb + r) * 65 + nG] = c[r] + bs + Vw[(mb + r) * 65 + nG];
      }
    }
    LDSF();
    {                                                // LN2(y) -> Hw f16
      float sm = 0.f, sq = 0.f;
      for (int j = 0; j < 32; ++j) { const float yv = Yw[t * 65 + fb + j]; sm += yv; sq += yv * yv; }
      sm += __shfl_xor(sm, 1); sq += __shfl_xor(sq, 1);
      const float mu = sm * (1.f / 64.f);
      const float rs = rsqrtf(sq * (1.f / 64.f) - mu * mu + 1e-5f);
      for (int j = 0; j < 32; ++j) {
        const int f = fb + j;
        Hw[t * 64 + f] = (_Float16)((Yw[t * 65 + f] - mu) * rs * g2[f] + b2[f]);
      }
    }
    LDSF();
    {                                                // z1 = gelu(z @ Wfc1^T + b) -> S
      const v16h a0 = loadA(Hw, lane, 0), a1 = loadA(Hw, lane, 1);
      for (int nt = 0; nt < 4; ++nt) {
        v8f c; for (int r = 0; r < 8; ++r) c[r] = 0.f;
        c = wmma32(a0, loadB(WF[28 + nt * 2 + 0], lane), c);
        c = wmma32(a1, loadB(WF[28 + nt * 2 + 1], lane), c);
        const int nG = nt * 16 + n; const float bs = bfc1[nG];
        for (int r = 0; r < 8; ++r) {
          float v = c[r] + bs;
          v = 0.5f * v * (1.f + erff(v * 0.70710678118654752f));  // exact GELU
          Sw[(mb + r) * 65 + nG] = v;
        }
      }
    }
    LDSF();
    for (int j = 0; j < 32; ++j)                     // z1 -> Hw f16
      Hw[t * 64 + fb + j] = (_Float16)Sw[t * 65 + fb + j];
    LDSF();
    {                                                // out = y + z1 @ Wfc2^T + b
      const v16h a0 = loadA(Hw, lane, 0), a1 = loadA(Hw, lane, 1);
      const size_t obase = ((size_t)b * LTOK + l0) * 64;
      for (int nt = 0; nt < 4; ++nt) {
        v8f c; for (int r = 0; r < 8; ++r) c[r] = 0.f;
        c = wmma32(a0, loadB(WF[36 + nt * 2 + 0], lane), c);
        c = wmma32(a1, loadB(WF[36 + nt * 2 + 1], lane), c);
        const int nG = nt * 16 + n; const float bs = bfc2[nG];
        for (int r = 0; r < 8; ++r)
          out[obase + (size_t)(mb + r) * 64 + nG] = c[r] + bs + Yw[(mb + r) * 65 + nG];
      }
    }
    LDSF();
  }
}

extern "C" void kernel_launch(void* const* d_in, const int* in_sizes, int n_in,
                              void* d_out, int out_size, void* d_ws, size_t ws_size,
                              hipStream_t stream) {
  const float* x     = (const float*)d_in[0];
  const float* ln1_g = (const float*)d_in[1];
  const float* ln1_b = (const float*)d_in[2];
  const float* w_kqv = (const float*)d_in[3];
  const float* b_kqv = (const float*)d_in[4];
  const float* w_prj = (const float*)d_in[5];
  const float* b_prj = (const float*)d_in[6];
  const float* ln2_g = (const float*)d_in[7];
  const float* ln2_b = (const float*)d_in[8];
  const float* w_fc1 = (const float*)d_in[9];
  const float* b_fc1 = (const float*)d_in[10];
  const float* w_fc2 = (const float*)d_in[11];
  const float* b_fc2 = (const float*)d_in[12];
  const float* w_rf  = (const float*)d_in[13];
  float* ws  = (float*)d_ws;
  float* out = (float*)d_out;
  (void)in_sizes; (void)n_in; (void)out_size; (void)ws_size;

  k1_reduce<<<BATCH * K1_BLK_PB, 128, 0, stream>>>(x, ln1_g, ln1_b, w_kqv, b_kqv, w_rf, ws);
  k_red<<<(BATCH * PART_STR + 255) / 256, 256, 0, stream>>>(ws);
  k2_finalize<<<BATCH * K2_BLK_PB, 128, 0, stream>>>(
      x, ln1_g, ln1_b, w_kqv, b_kqv, w_prj, b_prj, ln2_g, ln2_b,
      w_fc1, b_fc1, w_fc2, b_fc2, w_rf, ws, out);
}
